// TransformerBlock_72559177499412
// MI455X (gfx1250) — matrix operand order repaired and checked by reading, not compiled, not run on hardware
//
#include <hip/hip_runtime.h>

// ---------------------------------------------------------------------------
// Transformer block on MI455X (gfx1250, wave32). All GEMMs + attention run on
// v_wmma_f32_16x16x32_bf16 (bf16 inputs, fp32 accumulate).
// GEMM: 128x128 block tile, 4 waves, each wave 64x64 (16 WMMA tiles),
// software-pipelined global->reg->LDS staging.
// ---------------------------------------------------------------------------

typedef unsigned short u16;
typedef __attribute__((ext_vector_type(16))) __bf16 v16bf;
typedef __attribute__((ext_vector_type(8)))  float  v8f;

#define DEV static __device__ __forceinline__

// Model constants
constexpr int BATCH = 4;
constexpr int SEQ   = 2048;
constexpr int DMODEL= 1024;
constexpr int NHEAD = 16;
constexpr int HDIM  = 64;
constexpr int FFN   = 4096;
constexpr int MROWS = BATCH * SEQ;          // 8192

DEV u16 f32_to_bf16(float f) {
  unsigned u = __float_as_uint(f);
  u += 0x7FFFu + ((u >> 16) & 1u);          // round-to-nearest-even
  return (u16)(u >> 16);
}

DEV float gelu(float x) { return 0.5f * x * (1.0f + erff(x * 0.70710678118654752f)); }

union Frag {      // 16 bf16 operand fragment == 8 VGPRs
  uint4 u[2];
  v16bf v;
};

DEV v8f wmma_bf16(const Frag& a, const Frag& b, v8f c) {
  return __builtin_amdgcn_wmma_f32_16x16x32_bf16(false, a.v, false, b.v,
                                                 (short)0, c, false, false);
}

DEV v8f zero8() { v8f z = {0.f,0.f,0.f,0.f,0.f,0.f,0.f,0.f}; return z; }

// ---------------------------------------------------------------------------
// fp32 -> bf16 elementwise (for x)
// ---------------------------------------------------------------------------
__global__ __launch_bounds__(256) void cvt_kernel(const float* __restrict__ in,
                                                  u16* __restrict__ out, size_t n) {
  size_t i = ((size_t)blockIdx.x * 256 + threadIdx.x) * 4;
  if (i + 3 >= n) return;
  float4 f = *(const float4*)(in + i);
  u16 p[4] = { f32_to_bf16(f.x), f32_to_bf16(f.y), f32_to_bf16(f.z), f32_to_bf16(f.w) };
  *(uint2*)(out + i) = *(uint2*)p;
}

// ---------------------------------------------------------------------------
// fp32 [R][C] -> bf16 [C][R]  (weights into B-fragment-friendly [n][k] layout)
// ---------------------------------------------------------------------------
__global__ __launch_bounds__(256) void transpose_cvt_kernel(const float* __restrict__ in,
                                                            u16* __restrict__ out,
                                                            int R, int C) {
  __shared__ float tile[32][33];
  int c0 = blockIdx.x * 32, r0 = blockIdx.y * 32;
  #pragma unroll
  for (int i = 0; i < 4; i++)
    tile[threadIdx.y + i * 8][threadIdx.x] =
        in[(size_t)(r0 + threadIdx.y + i * 8) * C + c0 + threadIdx.x];
  __syncthreads();
  #pragma unroll
  for (int i = 0; i < 4; i++)
    out[(size_t)(c0 + threadIdx.y + i * 8) * R + r0 + threadIdx.x] =
        f32_to_bf16(tile[threadIdx.x][threadIdx.y + i * 8]);
}

// ---------------------------------------------------------------------------
// Tiled bf16 WMMA GEMM:  C[M][N] = A[M][K] * W[K][N] + bias, W given as Wt[N][K].
// Block: 128x128 tile, 128 threads (4 waves in a 2x2 grid), wave tile 64x64.
// Per K-step per wave: 16 WMMAs, 16 ds_load_b128 (1.0 b128/WMMA).
// Global loads for step k+1 are issued before the WMMAs of step k (latency hide).
// ---------------------------------------------------------------------------
enum { MQK = 0, MVT = 1, MORES = 2, MGELU = 3, MRES = 4 };

template <int MODE>
__global__ __launch_bounds__(128)
void gemm_kernel(const u16* __restrict__ A, const u16* __restrict__ Bt,
                 const float* __restrict__ bias, const float* __restrict__ resid,
                 float* __restrict__ outf, u16* __restrict__ outb,
                 int M, int N, int K) {
  __shared__ __align__(16) u16 lds_a[128][40];   // [m][k], padded row = 80B
  __shared__ __align__(16) u16 lds_b[128][40];   // [n][k]
  const int t    = threadIdx.x;
  const int m0   = blockIdx.y * 128;
  const int n0   = blockIdx.x * 128;
  const int w    = t >> 5, lane = t & 31;
  const int half = lane >> 4, lm = lane & 15;
  const int wm   = (w & 1) * 64;   // wave row offset within tile
  const int wn   = (w >> 1) * 64;  // wave col offset within tile

  v8f acc[4][4];
  #pragma unroll
  for (int i = 0; i < 4; i++)
    #pragma unroll
    for (int j = 0; j < 4; j++) acc[i][j] = zero8();

  // Each thread owns one 128-row tile row: 4 x b128 per matrix per K-step.
  const u16* ga = A  + (size_t)(m0 + t) * K;
  const u16* gb = Bt + (size_t)(n0 + t) * K;
  uint4 pa[4], pb[4];
  #pragma unroll
  for (int u = 0; u < 4; u++) {            // prefetch tile 0
    pa[u] = *(const uint4*)(ga + u * 8);
    pb[u] = *(const uint4*)(gb + u * 8);
  }

  for (int kb = 0; kb < K; kb += 32) {
    __syncthreads();                       // prior step's LDS reads done
    #pragma unroll
    for (int u = 0; u < 4; u++) {
      *(uint4*)&lds_a[t][u * 8] = pa[u];
      *(uint4*)&lds_b[t][u * 8] = pb[u];
    }
    __syncthreads();
    if (kb + 32 < K) {                     // prefetch next tile; overlaps WMMAs
      #pragma unroll
      for (int u = 0; u < 4; u++) {
        pa[u] = *(const uint4*)(ga + kb + 32 + u * 8);
        pb[u] = *(const uint4*)(gb + kb + 32 + u * 8);
      }
    }

    Frag a[4];
    #pragma unroll
    for (int i = 0; i < 4; i++) {          // A frag: K = half*8 .. , 16+half*8 ..
      int r = wm + i * 16 + lm;
      a[i].u[0] = *(const uint4*)&lds_a[r][half * 8];
      a[i].u[1] = *(const uint4*)&lds_a[r][16 + half * 8];
    }
    #pragma unroll
    for (int j = 0; j < 4; j++) {          // B frag: K = half*8 .. , 16+half*8 ..
      Frag b;
      int r = wn + j * 16 + lm;
      b.u[0] = *(const uint4*)&lds_b[r][half * 8];
      b.u[1] = *(const uint4*)&lds_b[r][16 + half * 8];
      #pragma unroll
      for (int i = 0; i < 4; i++)
        acc[i][j] = wmma_bf16(a[i], b, acc[i][j]);
    }
  }

  // Epilogue. C layout: VGPR v -> row (v + 8*half), col lm within 16x16 tile.
  #pragma unroll
  for (int i = 0; i < 4; i++) {
    #pragma unroll
    for (int j = 0; j < 4; j++) {
      const int n = n0 + wn + j * 16 + lm;
      const int mbase = m0 + wm + i * 16 + half * 8;
      const float bn = bias[n];
      if constexpr (MODE == MQK) {
        // bf16 out at [b][h][s][hd]
        const int bI = mbase >> 11, h = n >> 6, hd = n & 63, s0 = mbase & 2047;
        u16* p = outb + ((size_t)(bI * NHEAD + h) * SEQ + s0) * HDIM + hd;
        #pragma unroll
        for (int v = 0; v < 8; v++) p[(size_t)v * HDIM] = f32_to_bf16(acc[i][j][v] + bn);
      } else if constexpr (MODE == MVT) {
        // bf16 out at [b][h][hd][s]  (pre-transposed V): 8 consecutive s per lane
        const int bI = mbase >> 11, h = n >> 6, hd = n & 63, s0 = mbase & 2047;
        __align__(16) u16 pk[8];
        #pragma unroll
        for (int v = 0; v < 8; v++) pk[v] = f32_to_bf16(acc[i][j][v] + bn);
        *(uint4*)(outb + ((size_t)(bI * NHEAD + h) * HDIM + hd) * SEQ + s0) = *(uint4*)pk;
      } else if constexpr (MODE == MORES || MODE == MRES) {
        // fp32 out + residual
        #pragma unroll
        for (int v = 0; v < 8; v++) {
          size_t o = (size_t)(mbase + v) * N + n;
          outf[o] = acc[i][j][v] + bn + resid[o];
        }
      } else { // MGELU: bf16 gelu(acc+bias)
        #pragma unroll
        for (int v = 0; v < 8; v++)
          outb[(size_t)(mbase + v) * N + n] = f32_to_bf16(gelu(acc[i][j][v] + bn));
      }
    }
  }
}

// ---------------------------------------------------------------------------
// Flash attention. Block = 128 threads (4 waves) = 64 query rows of one (b,h).
// Wave handles 16 rows; loops over 64-key blocks with online softmax.
// q*K^T: A = q (regs), B = K tile in LDS[key][hd].
// P*V  : A = P (LDS round-trip in A-frag layout), B = Vt tile in LDS[hd][key].
// K/V/mask tiles for block kb+1 prefetched into regs during block kb's compute.
// ---------------------------------------------------------------------------
__global__ __launch_bounds__(128)
void attn_kernel(const u16* __restrict__ qg, const u16* __restrict__ kg,
                 const u16* __restrict__ vtg, const int* __restrict__ mask,
                 u16* __restrict__ ctxb) {
  __shared__ __align__(16) u16 lds_k[64][72];
  __shared__ __align__(16) u16 lds_v[64][72];
  __shared__ __align__(16) u16 lds_p[4][16][72];
  __shared__ float lds_mask[64];

  const int t = threadIdx.x;
  const int w = t >> 5, lane = t & 31, half = lane >> 4, lm = lane & 15;
  const int bh = blockIdx.y, bI = bh >> 4;
  const int sblk = blockIdx.x * 64;
  const size_t head_off = (size_t)bh * SEQ * HDIM;

  // q fragments for this wave's 16 rows (K = hd = 64 -> 2 chunks of 32)
  Frag qa[2];
  {
    const u16* qrow = qg + head_off + (size_t)(sblk + w * 16 + lm) * HDIM;
    #pragma unroll
    for (int c = 0; c < 2; c++) {
      qa[c].u[0] = *(const uint4*)(qrow + c * 32 + half * 8);
      qa[c].u[1] = *(const uint4*)(qrow + c * 32 + 16 + half * 8);
    }
  }

  v8f ctx[4];
  #pragma unroll
  for (int jh = 0; jh < 4; jh++) ctx[jh] = zero8();
  float rmax[8], rsum[8];
  #pragma unroll
  for (int v = 0; v < 8; v++) { rmax[v] = -INFINITY; rsum[v] = 0.f; }

  // tile row ownership: idx = t*4+u -> row = idx>>3, chunk q = idx&7
  uint4 pk[4], pv[4];
  float pm = 0.f;
  #pragma unroll
  for (int u = 0; u < 4; u++) {            // prefetch key-block 0
    int idx = t * 4 + u, row = idx >> 3, q = idx & 7;
    pk[u] = *(const uint4*)(kg + head_off + (size_t)row * HDIM + q * 8);
    pv[u] = *(const uint4*)(vtg + head_off + (size_t)row * SEQ + q * 8);
  }
  if (t < 64) pm = (float)mask[bI * SEQ + t] * -4294967296.0f;

  for (int kblk = 0; kblk < SEQ; kblk += 64) {
    __syncthreads();
    #pragma unroll
    for (int u = 0; u < 4; u++) {
      int idx = t * 4 + u, row = idx >> 3, q = idx & 7;
      *(uint4*)&lds_k[row][q * 8] = pk[u];
      *(uint4*)&lds_v[row][q * 8] = pv[u];
    }
    if (t < 64) lds_mask[t] = pm;
    __syncthreads();
    if (kblk + 64 < SEQ) {                 // prefetch next key block
      #pragma unroll
      for (int u = 0; u < 4; u++) {
        int idx = t * 4 + u, row = idx >> 3, q = idx & 7;
        pk[u] = *(const uint4*)(kg + head_off + (size_t)(kblk + 64 + row) * HDIM + q * 8);
        pv[u] = *(const uint4*)(vtg + head_off + (size_t)row * SEQ + kblk + 64 + q * 8);
      }
      if (t < 64) pm = (float)mask[bI * SEQ + kblk + 64 + t] * -4294967296.0f;
    }

    // logits tile 16x64 for this wave
    v8f l[4];
    #pragma unroll
    for (int j = 0; j < 4; j++) l[j] = zero8();
    #pragma unroll
    for (int c = 0; c < 2; c++) {
      #pragma unroll
      for (int j = 0; j < 4; j++) {
        Frag fb;
        int r = j * 16 + lm;
        fb.u[0] = *(const uint4*)&lds_k[r][c * 32 + half * 8];
        fb.u[1] = *(const uint4*)&lds_k[r][c * 32 + 16 + half * 8];
        l[j] = wmma_bf16(qa[c], fb, l[j]);
      }
    }

    // online softmax per row (row = v + 8*half lives across lanes lm=0..15)
    float p[4][8];
    #pragma unroll
    for (int v = 0; v < 8; v++) {
      float mx = -INFINITY;
      #pragma unroll
      for (int j = 0; j < 4; j++) {
        float val = l[j][v] * 0.125f + lds_mask[j * 16 + lm];
        l[j][v] = val;
        mx = fmaxf(mx, val);
      }
      #pragma unroll
      for (int d = 1; d < 16; d <<= 1) mx = fmaxf(mx, __shfl_xor(mx, d, 32));
      float nm = fmaxf(rmax[v], mx);
      float corr = __expf(rmax[v] - nm);
      rmax[v] = nm;
      rsum[v] *= corr;
      #pragma unroll
      for (int jh = 0; jh < 4; jh++) ctx[jh][v] *= corr;
      float ls = 0.f;
      #pragma unroll
      for (int j = 0; j < 4; j++) {
        float e = __expf(l[j][v] - nm);
        p[j][v] = e;
        ls += e;
      }
      #pragma unroll
      for (int d = 1; d < 16; d <<= 1) ls += __shfl_xor(ls, d, 32);
      rsum[v] += ls;
    }

    // stage P (bf16) into per-wave LDS region, [m][key] for A-fragment reads
    #pragma unroll
    for (int j = 0; j < 4; j++)
      #pragma unroll
      for (int v = 0; v < 8; v++)
        lds_p[w][half * 8 + v][j * 16 + lm] = f32_to_bf16(p[j][v]);
    asm volatile("s_wait_dscnt 0" ::: "memory");   // same-wave cross-lane LDS RAW

    // ctx += P (16x64) * V (64x64)
    #pragma unroll
    for (int c = 0; c < 2; c++) {
      Frag pa;
      pa.u[0] = *(const uint4*)&lds_p[w][lm][c * 32 + half * 8];
      pa.u[1] = *(const uint4*)&lds_p[w][lm][c * 32 + 16 + half * 8];
      #pragma unroll
      for (int jh = 0; jh < 4; jh++) {
        Frag fb;
        int r = jh * 16 + lm;
        fb.u[0] = *(const uint4*)&lds_v[r][c * 32 + half * 8];
        fb.u[1] = *(const uint4*)&lds_v[r][c * 32 + 16 + half * 8];
        ctx[jh] = wmma_bf16(pa, fb, ctx[jh]);
      }
    }
  }

  // normalize + write ctx as bf16 row-major [m][D] for the Wo GEMM
  #pragma unroll
  for (int v = 0; v < 8; v++) {
    float inv = 1.0f / rsum[v];
    int s = sblk + w * 16 + half * 8 + v;
    size_t base = ((size_t)bI * SEQ + s) * DMODEL + (bh & 15) * HDIM;
    #pragma unroll
    for (int jh = 0; jh < 4; jh++)
      ctxb[base + jh * 16 + lm] = f32_to_bf16(ctx[jh][v] * inv);
  }
}

// ---------------------------------------------------------------------------
// LayerNorm over D=1024, one block (256 thr) per row. Optionally emits bf16.
// ---------------------------------------------------------------------------
template <bool WB16>
__global__ __launch_bounds__(256)
void ln_kernel(const float* __restrict__ in, const float* __restrict__ gamma,
               const float* __restrict__ beta, float* __restrict__ out32,
               u16* __restrict__ out16) {
  __shared__ float ps[8], pq[8];
  const int t = threadIdx.x, w = t >> 5, lane = t & 31;
  const size_t row = blockIdx.x;
  float vals[4], s = 0.f, q = 0.f;
  #pragma unroll
  for (int i = 0; i < 4; i++) {
    float v = in[row * DMODEL + t + i * 256];
    vals[i] = v; s += v; q += v * v;
  }
  #pragma unroll
  for (int d = 1; d < 32; d <<= 1) { s += __shfl_xor(s, d, 32); q += __shfl_xor(q, d, 32); }
  if (lane == 0) { ps[w] = s; pq[w] = q; }
  __syncthreads();
  float ts = 0.f, tq = 0.f;
  #pragma unroll
  for (int i = 0; i < 8; i++) { ts += ps[i]; tq += pq[i]; }
  const float mu = ts * (1.0f / DMODEL);
  const float var = tq * (1.0f / DMODEL) - mu * mu;
  const float rs = rsqrtf(var + 1e-5f);
  #pragma unroll
  for (int i = 0; i < 4; i++) {
    int n = t + i * 256;
    float y = (vals[i] - mu) * rs * gamma[n] + beta[n];
    out32[row * DMODEL + n] = y;
    if constexpr (WB16) out16[row * DMODEL + n] = f32_to_bf16(y);
  }
}

// ---------------------------------------------------------------------------
// Host-side orchestration
// ---------------------------------------------------------------------------
extern "C" void kernel_launch(void* const* d_in, const int* in_sizes, int n_in,
                              void* d_out, int out_size, void* d_ws, size_t ws_size,
                              hipStream_t stream) {
  (void)in_sizes; (void)n_in; (void)out_size;
  const float* x     = (const float*)d_in[0];
  const int*   mask  = (const int*)d_in[1];
  const float* Wq    = (const float*)d_in[2];
  const float* bq    = (const float*)d_in[3];
  const float* Wk    = (const float*)d_in[4];
  const float* bk    = (const float*)d_in[5];
  const float* Wv    = (const float*)d_in[6];
  const float* bv    = (const float*)d_in[7];
  const float* Wo    = (const float*)d_in[8];
  const float* bo    = (const float*)d_in[9];
  const float* ln1_s = (const float*)d_in[10];
  const float* ln1_b = (const float*)d_in[11];
  const float* W1    = (const float*)d_in[12];
  const float* b1    = (const float*)d_in[13];
  const float* W2    = (const float*)d_in[14];
  const float* b2    = (const float*)d_in[15];
  const float* ln2_s = (const float*)d_in[16];
  const float* ln2_b = (const float*)d_in[17];

  const size_t MB = 1024 * 1024;
  if (ws_size < 184 * MB) return;
  char* ws = (char*)d_ws;
  // persistent bf16 weights [n][k]
  u16* Wqt = (u16*)(ws + 0 * MB);
  u16* Wkt = (u16*)(ws + 2 * MB);
  u16* Wvt = (u16*)(ws + 4 * MB);
  u16* Wot = (u16*)(ws + 6 * MB);
  u16* W1t = (u16*)(ws + 8 * MB);
  u16* W2t = (u16*)(ws + 16 * MB);
  // phase 1 region (reused by `act` after attention path completes)
  u16* xb  = (u16*)(ws + 24 * MB);
  u16* qb  = (u16*)(ws + 40 * MB);
  u16* kb  = (u16*)(ws + 56 * MB);
  u16* vtb = (u16*)(ws + 72 * MB);
  u16* act = (u16*)(ws + 24 * MB);   // 64 MB, overlays xb..vtb
  u16* ctxb = (u16*)(ws + 88 * MB);
  float* attn_pre = (float*)(ws + 104 * MB);  // reused as mlp_pre
  float* mlp_pre  = (float*)(ws + 104 * MB);
  float* h  = (float*)(ws + 136 * MB);
  u16*   hb = (u16*)(ws + 168 * MB);

  const dim3 tb(32, 8);
  // weights -> bf16 [n][k]
  transpose_cvt_kernel<<<dim3(32, 32), tb, 0, stream>>>(Wq, Wqt, DMODEL, DMODEL);
  transpose_cvt_kernel<<<dim3(32, 32), tb, 0, stream>>>(Wk, Wkt, DMODEL, DMODEL);
  transpose_cvt_kernel<<<dim3(32, 32), tb, 0, stream>>>(Wv, Wvt, DMODEL, DMODEL);
  transpose_cvt_kernel<<<dim3(32, 32), tb, 0, stream>>>(Wo, Wot, DMODEL, DMODEL);
  transpose_cvt_kernel<<<dim3(128, 32), tb, 0, stream>>>(W1, W1t, DMODEL, FFN);
  transpose_cvt_kernel<<<dim3(32, 128), tb, 0, stream>>>(W2, W2t, FFN, DMODEL);
  // x -> bf16
  cvt_kernel<<<(MROWS * DMODEL) / 1024, 256, 0, stream>>>(x, xb, (size_t)MROWS * DMODEL);

  // QKV projections (all wmma)
  gemm_kernel<MQK><<<dim3(8, 64), 128, 0, stream>>>(xb, Wqt, bq, nullptr, nullptr, qb,
                                                    MROWS, DMODEL, DMODEL);
  gemm_kernel<MQK><<<dim3(8, 64), 128, 0, stream>>>(xb, Wkt, bk, nullptr, nullptr, kb,
                                                    MROWS, DMODEL, DMODEL);
  gemm_kernel<MVT><<<dim3(8, 64), 128, 0, stream>>>(xb, Wvt, bv, nullptr, nullptr, vtb,
                                                    MROWS, DMODEL, DMODEL);
  // attention
  attn_kernel<<<dim3(SEQ / 64, BATCH * NHEAD), 128, 0, stream>>>(qb, kb, vtb, mask, ctxb);
  // output projection + residual (fp32)
  gemm_kernel<MORES><<<dim3(8, 64), 128, 0, stream>>>(ctxb, Wot, bo, x, attn_pre, nullptr,
                                                      MROWS, DMODEL, DMODEL);
  // LN1 -> h (fp32) + hb (bf16)
  ln_kernel<true><<<MROWS, 256, 0, stream>>>(attn_pre, ln1_s, ln1_b, h, hb);
  // MLP
  gemm_kernel<MGELU><<<dim3(32, 64), 128, 0, stream>>>(hb, W1t, b1, nullptr, nullptr, act,
                                                       MROWS, FFN, DMODEL);
  gemm_kernel<MRES><<<dim3(8, 64), 128, 0, stream>>>(act, W2t, b2, h, mlp_pre, nullptr,
                                                     MROWS, DMODEL, FFN);
  // LN2 -> final fp32 output
  ln_kernel<false><<<MROWS, 256, 0, stream>>>(mlp_pre, ln2_s, ln2_b, (float*)d_out, nullptr);
}
